// SampledCenterLoss_4226247819399
// MI455X (gfx1250) — matrix operand order repaired and checked by reading, not compiled, not run on hardware
//
#include <hip/hip_runtime.h>
#include <hip/hip_bf16.h>

#define BN 4096
#define CN 128
#define DN 768
#define MAXM 128
#define MAXP (MAXM * (MAXM - 1) / 2) /* 8128 */
#define EPSF 1e-12f
#define NSORT 8192
#define HHALF 8388608u /* BN*BN/2 */

typedef __attribute__((ext_vector_type(16))) __bf16 v16bf;
typedef __attribute__((ext_vector_type(8))) float v8f;

// ---------------------------------------------------------------- utilities

__device__ __forceinline__ unsigned rotl32(unsigned x, int r) {
  return (x << r) | (x >> (32 - r));
}

// Exact reproduction of jax.random.uniform(jax.random.key(1), (B,B), f32)
// at flat index idx = i*BN + j. Threefry2x32 with key (0,1); counters are
// iota split into halves (p, p+H) -> (out0[p], out1[p]).
__device__ __forceinline__ float jax_uniform01(unsigned idx) {
  unsigned c0, c1; int pick;
  if (idx < HHALF) { c0 = idx; c1 = idx + HHALF; pick = 0; }
  else             { c0 = idx - HHALF; c1 = idx; pick = 1; }
  const unsigned ks0 = 0u, ks1 = 1u;
  const unsigned ks2 = 0x1BD11BDAu ^ ks0 ^ ks1;
  unsigned x0 = c0 + ks0, x1 = c1 + ks1;
  const int RA[4] = {13, 15, 26, 6};
  const int RB[4] = {17, 29, 16, 24};
#pragma unroll
  for (int i = 0; i < 4; ++i) { x0 += x1; x1 = rotl32(x1, RA[i]); x1 ^= x0; }
  x0 += ks1; x1 += ks2 + 1u;
#pragma unroll
  for (int i = 0; i < 4; ++i) { x0 += x1; x1 = rotl32(x1, RB[i]); x1 ^= x0; }
  x0 += ks2; x1 += ks0 + 2u;
#pragma unroll
  for (int i = 0; i < 4; ++i) { x0 += x1; x1 = rotl32(x1, RA[i]); x1 ^= x0; }
  x0 += ks0; x1 += ks1 + 3u;
#pragma unroll
  for (int i = 0; i < 4; ++i) { x0 += x1; x1 = rotl32(x1, RB[i]); x1 ^= x0; }
  x0 += ks1; x1 += ks2 + 4u;
#pragma unroll
  for (int i = 0; i < 4; ++i) { x0 += x1; x1 = rotl32(x1, RA[i]); x1 ^= x0; }
  x0 += ks2; x1 += ks0 + 5u;
  unsigned bits = pick ? x1 : x0;
  return __uint_as_float((bits >> 9) | 0x3F800000u) - 1.0f;
}

// ---------------------------------------------------------------- kernels

// Rows [0,BN): features -> E.  Rows [BN, BN+CN): centers -> Cn.
__global__ __launch_bounds__(256) void k_normalize(
    const float* __restrict__ feat, const float* __restrict__ cent,
    float* __restrict__ E, float* __restrict__ Cn) {
  int row = blockIdx.x, tid = threadIdx.x;
  const float* src; float* dst;
  if (row < BN) { src = feat + (size_t)row * DN; dst = E + (size_t)row * DN; }
  else { src = cent + (size_t)(row - BN) * DN; dst = Cn + (size_t)(row - BN) * DN; }
  float s = 0.f;
  for (int k = tid; k < DN; k += 256) { float v = src[k]; s += v * v; }
  __shared__ float red[256];
  red[tid] = s; __syncthreads();
  for (int off = 128; off > 0; off >>= 1) {
    if (tid < off) red[tid] += red[tid + off];
    __syncthreads();
  }
  float d = fmaxf(sqrtf(red[0]), EPSF);
  for (int k = tid; k < DN; k += 256) dst[k] = src[k] / d;
}

// Per-class member count via deterministic wave32 ballot scan.
__global__ void k_count(const int* __restrict__ labels, int* __restrict__ m) {
  int c = blockIdx.x, lane = threadIdx.x;
  int cnt = 0;
  for (int i0 = 0; i0 < BN; i0 += 32) {
    unsigned msk = (unsigned)__ballot(labels[i0 + lane] == c);
    cnt += __popc(msk);
  }
  if (lane == 0) m[c] = cnt;
}

__global__ void k_prefix(const int* __restrict__ m, int* __restrict__ cstart) {
  if (threadIdx.x == 0 && blockIdx.x == 0) {
    int acc = 0;
    for (int c = 0; c < CN; ++c) {
      cstart[c] = acc;
      int mc = m[c]; if (mc > MAXM) mc = MAXM;
      acc += mc;
    }
  }
}

// Deterministic stable scatter: rows of class c in increasing original index.
__global__ void k_scatter(const int* __restrict__ labels, const int* __restrict__ m,
                          const int* __restrict__ cstart, int* __restrict__ rows) {
  int c = blockIdx.x, lane = threadIdx.x;
  int base = cstart[c];
  int cap = m[c] < MAXM ? m[c] : MAXM;
  int cnt = 0;
  for (int i0 = 0; i0 < BN; i0 += 32) {
    int i = i0 + lane;
    bool hit = (labels[i] == c);
    unsigned msk = (unsigned)__ballot(hit);
    int rank = __popc(msk & ((1u << lane) - 1u));
    if (hit && (cnt + rank) < cap) rows[base + cnt + rank] = i;
    cnt += __popc(msk);
  }
}

// EC[i] = dot(E[i], Cn[labels[i]])
__global__ __launch_bounds__(256) void k_ec(
    const float* __restrict__ E, const float* __restrict__ Cn,
    const int* __restrict__ labels, float* __restrict__ EC) {
  int i = blockIdx.x, tid = threadIdx.x;
  const float* e = E + (size_t)i * DN;
  const float* cc = Cn + (size_t)labels[i] * DN;
  float s = 0.f;
  for (int k = tid; k < DN; k += 256) s += e[k] * cc[k];
  __shared__ float red[256];
  red[tid] = s; __syncthreads();
  for (int off = 128; off > 0; off >>= 1) {
    if (tid < off) red[tid] += red[tid + off];
    __syncthreads();
  }
  if (tid == 0) EC[i] = red[0];
}

// Per-class Gram via WMMA (bf16x3 split: hi*hi + hi*lo + lo*hi ~ f32 accuracy).
// One workgroup (8 waves) per class; each wave owns 16x16 output tiles.
__global__ __launch_bounds__(256) void k_gram(
    const float* __restrict__ E, const int* __restrict__ m,
    const int* __restrict__ cstart, const int* __restrict__ rows,
    float* __restrict__ pairS, int* __restrict__ pairAB) {
  int c = blockIdx.x;
  int Mc = m[c] < MAXM ? m[c] : MAXM;
  if (Mc < 2) return;
  __shared__ int rs[MAXM];
  int tid = threadIdx.x;
  int base = cstart[c];
  if (tid < MAXM) rs[tid] = (tid < Mc) ? rows[base + tid] : 0;
  __syncthreads();

  int nt = (Mc + 15) >> 4;
  int ntiles = nt * (nt + 1) / 2;          // upper-triangular tile set ti<=tj
  int w = tid >> 5, lane = tid & 31;
  int h = lane >> 4, mm = lane & 15;
  size_t pbase = (size_t)c * MAXP;

  for (int t = w; t < ntiles; t += 8) {
    int ti = 0, rem = t;
    while (rem >= nt - ti) { rem -= nt - ti; ++ti; }
    int tj = ti + rem;

    int ra = ti * 16 + mm, rb = tj * 16 + mm;
    const float* pa = (ra < Mc) ? (E + (size_t)rs[ra] * DN) : nullptr;
    const float* pb = (rb < Mc) ? (E + (size_t)rs[rb] * DN) : nullptr;

    v8f acc = {0.f, 0.f, 0.f, 0.f, 0.f, 0.f, 0.f, 0.f};
    for (int k0 = 0; k0 < DN; k0 += 32) {
      float fa[16], fb[16];
      // A (16x32 bf16 layout): lanes 0-15 -> K k0+0..7 / k0+16..23,
      //                        lanes 16-31 -> K k0+8..15 / k0+24..31
      if (pa) {
        *(float4*)&fa[0]  = *(const float4*)(pa + k0 + h * 8);
        *(float4*)&fa[4]  = *(const float4*)(pa + k0 + h * 8 + 4);
        *(float4*)&fa[8]  = *(const float4*)(pa + k0 + 16 + h * 8);
        *(float4*)&fa[12] = *(const float4*)(pa + k0 + 16 + h * 8 + 4);
      } else {
#pragma unroll
        for (int v = 0; v < 16; ++v) fa[v] = 0.f;
      }
      // B (32x16 bf16 layout): N = lane&15; lanes 0-15 hold K k0+0..7 / k0+16..23,
      //                        lanes 16-31 hold K k0+8..15 / k0+24..31
      if (pb) {
        *(float4*)&fb[0]  = *(const float4*)(pb + k0 + h * 8);
        *(float4*)&fb[4]  = *(const float4*)(pb + k0 + h * 8 + 4);
        *(float4*)&fb[8]  = *(const float4*)(pb + k0 + 16 + h * 8);
        *(float4*)&fb[12] = *(const float4*)(pb + k0 + 16 + h * 8 + 4);
      } else {
#pragma unroll
        for (int v = 0; v < 16; ++v) fb[v] = 0.f;
      }
      v16bf ahi, alo, bhi, blo;
#pragma unroll
      for (int v = 0; v < 16; ++v) {
        __bf16 hh = (__bf16)fa[v];
        ahi[v] = hh; alo[v] = (__bf16)(fa[v] - (float)hh);
        __bf16 gg = (__bf16)fb[v];
        bhi[v] = gg; blo[v] = (__bf16)(fb[v] - (float)gg);
      }
      acc = __builtin_amdgcn_wmma_f32_16x16x32_bf16(false, ahi, false, bhi,
                                                    (short)0, acc, false, false);
      acc = __builtin_amdgcn_wmma_f32_16x16x32_bf16(false, ahi, false, blo,
                                                    (short)0, acc, false, false);
      acc = __builtin_amdgcn_wmma_f32_16x16x32_bf16(false, alo, false, bhi,
                                                    (short)0, acc, false, false);
    }
    // C/D layout: VGPR r holds M = r + 8*h, N = lane&15
#pragma unroll
    for (int r = 0; r < 8; ++r) {
      int a = ti * 16 + r + 8 * h;
      int b = tj * 16 + mm;
      if (a < Mc && b < Mc && a < b) {
        int pidx = a * Mc - (a * (a + 1)) / 2 + (b - a - 1);
        pairS[pbase + pidx] = acc[r];
        pairAB[pbase + pidx] = (a << 16) | b;
      }
    }
  }
}

// Per-class bitonic sort of pd values in LDS; thr[c] = pd_sorted[(np-1)/2].
__global__ __launch_bounds__(256) void k_median(
    const float* __restrict__ pairS, const int* __restrict__ m,
    float* __restrict__ thr) {
  __shared__ float sb[NSORT];
  int c = blockIdx.x, tid = threadIdx.x;
  int Mc = m[c] < MAXM ? m[c] : MAXM;
  int np = Mc * (Mc - 1) / 2;
  size_t pbase = (size_t)c * MAXP;
  for (int p = tid; p < NSORT; p += 256) {
    float v = __builtin_inff();
    if (p < np) {
      float s = pairS[pbase + p];
      v = 1.f - fminf(fmaxf(s, -1.f), 1.f);
    }
    sb[p] = v;
  }
  __syncthreads();
  for (int ksz = 2; ksz <= NSORT; ksz <<= 1) {
    for (int jsz = ksz >> 1; jsz > 0; jsz >>= 1) {
      for (int t2 = tid; t2 < NSORT; t2 += 256) {
        int ixj = t2 ^ jsz;
        if (ixj > t2) {
          float x = sb[t2], y = sb[ixj];
          bool up = ((t2 & ksz) == 0);
          if (up ? (x > y) : (x < y)) { sb[t2] = y; sb[ixj] = x; }
        }
      }
      __syncthreads();
    }
  }
  if (tid == 0) thr[c] = (np > 0) ? sb[(np - 1) >> 1] : __builtin_inff();
}

// Selected-pair loss: threefry r on demand, deterministic block reduction.
__global__ __launch_bounds__(256) void k_loss(
    const float* __restrict__ pairS, const int* __restrict__ pairAB,
    const int* __restrict__ m, const int* __restrict__ cstart,
    const int* __restrict__ rows, const float* __restrict__ thr,
    const float* __restrict__ EC,
    float* __restrict__ classLoss, float* __restrict__ classW) {
  int c = blockIdx.x, tid = threadIdx.x;
  int Mc = m[c] < MAXM ? m[c] : MAXM;
  int np = Mc * (Mc - 1) / 2;
  float tc = thr[c];
  float wgt = (float)m[c];
  size_t pbase = (size_t)c * MAXP;
  int base = cstart[c];
  float lsum = 0.f, wsum = 0.f;
  for (int p = tid; p < np; p += 256) {
    float s = pairS[pbase + p];
    float sclip = fminf(fmaxf(s, -1.f), 1.f);
    float pd = 1.f - sclip;
    if (pd > tc) {
      int ab = pairAB[pbase + p];
      int a = ab >> 16, b = ab & 0xffff;
      int i = rows[base + a], j = rows[base + b];   // i < j guaranteed
      float r = jax_uniform01((unsigned)i * (unsigned)BN + (unsigned)j);
      float one_r = 1.f - r;
      float norm2 = r * r + one_r * one_r + 2.f * r * one_r * s;  // raw S
      float nrm = fmaxf(sqrtf(fmaxf(norm2, 0.f)), EPSF);
      float dot = (r * EC[i] + one_r * EC[j]) / nrm;
      dot = fminf(fmaxf(dot, -1.f), 1.f);
      lsum += wgt * (1.f - dot);
      wsum += wgt;
    }
  }
  __shared__ float rl[256], rw[256];
  rl[tid] = lsum; rw[tid] = wsum; __syncthreads();
  for (int off = 128; off > 0; off >>= 1) {
    if (tid < off) { rl[tid] += rl[tid + off]; rw[tid] += rw[tid + off]; }
    __syncthreads();
  }
  if (tid == 0) { classLoss[c] = rl[0]; classW[c] = rw[0]; }
}

__global__ void k_final(const float* __restrict__ classLoss,
                        const float* __restrict__ classW,
                        float* __restrict__ out) {
  if (threadIdx.x == 0 && blockIdx.x == 0) {
    float L = 0.f, W = 0.f;
    for (int c = 0; c < CN; ++c) { L += classLoss[c]; W += classW[c]; }
    out[0] = (W > 0.f) ? (L / W) : 0.f;
  }
}

// ---------------------------------------------------------------- launch

extern "C" void kernel_launch(void* const* d_in, const int* in_sizes, int n_in,
                              void* d_out, int out_size, void* d_ws, size_t ws_size,
                              hipStream_t stream) {
  (void)in_sizes; (void)n_in; (void)out_size; (void)ws_size;
  const float* feat = (const float*)d_in[0];
  const float* cent = (const float*)d_in[1];
  const int* labels = (const int*)d_in[2];
  // d_in[3] = cam_ids: unused by the reference forward value.

  char* ws = (char*)d_ws;
  size_t off = 0;
  auto alloc = [&](size_t bytes) -> char* {
    char* p = ws + off;
    off += (bytes + 255) & ~(size_t)255;
    return p;
  };
  float* E         = (float*)alloc((size_t)BN * DN * sizeof(float));
  float* Cn        = (float*)alloc((size_t)CN * DN * sizeof(float));
  float* EC        = (float*)alloc((size_t)BN * sizeof(float));
  float* thr       = (float*)alloc((size_t)CN * sizeof(float));
  float* classLoss = (float*)alloc((size_t)CN * sizeof(float));
  float* classW    = (float*)alloc((size_t)CN * sizeof(float));
  float* pairS     = (float*)alloc((size_t)CN * MAXP * sizeof(float));
  int*   mcnt      = (int*)alloc((size_t)CN * sizeof(int));
  int*   cstart    = (int*)alloc((size_t)CN * sizeof(int));
  int*   rows      = (int*)alloc((size_t)BN * sizeof(int));
  int*   pairAB    = (int*)alloc((size_t)CN * MAXP * sizeof(int));

  k_normalize<<<BN + CN, 256, 0, stream>>>(feat, cent, E, Cn);
  k_count<<<CN, 32, 0, stream>>>(labels, mcnt);
  k_prefix<<<1, 1, 0, stream>>>(mcnt, cstart);
  k_scatter<<<CN, 32, 0, stream>>>(labels, mcnt, cstart, rows);
  k_ec<<<BN, 256, 0, stream>>>(E, Cn, labels, EC);
  k_gram<<<CN, 256, 0, stream>>>(E, mcnt, cstart, rows, pairS, pairAB);
  k_median<<<CN, 256, 0, stream>>>(pairS, mcnt, thr);
  k_loss<<<CN, 256, 0, stream>>>(pairS, pairAB, mcnt, cstart, rows, thr, EC,
                                 classLoss, classW);
  k_final<<<1, 1, 0, stream>>>(classLoss, classW, (float*)d_out);
}
